// LRUBlock_61065845015259
// MI455X (gfx1250) — matrix operand order repaired and checked by reading, not compiled, not run on hardware
//
#include <hip/hip_runtime.h>
#include <hip/hip_bf16.h>
#include <math.h>

// ---------------------------------------------------------------------------
// Problem constants (match reference)
// ---------------------------------------------------------------------------
#define BB 4
#define TT 4096
#define II 512
#define HH 512
#define LL 4
#define FF (4 * HH)        // 2048
#define MM (BB * TT)       // 16384 rows (tokens)

// ---------------------------------------------------------------------------
// CDNA5 WMMA types
// ---------------------------------------------------------------------------
typedef __attribute__((ext_vector_type(16))) __bf16        v16bf;
typedef __attribute__((ext_vector_type(8)))  float         v8f;
typedef __attribute__((ext_vector_type(4)))  unsigned int  u32x4;

union Frag  { v16bf v; u32x4 u[2]; };
union AccSp { v8f   v; u32x4 q[2]; };

__device__ __forceinline__ v8f wmma_bf16(v16bf a, v16bf b, v8f c) {
  // D = A(16x32 bf16) * B(32x16 bf16) + C(16x16 f32)
  return __builtin_amdgcn_wmma_f32_16x16x32_bf16(
      /*neg_a=*/false, a, /*neg_b=*/false, b,
      /*c_mod=*/(short)0, c, /*reuse_a=*/false, /*reuse_b=*/false);
}

// ---------------------------------------------------------------------------
// Elementwise: f32 -> bf16 cast with optional per-row exp(gamma) scale & mul
// ---------------------------------------------------------------------------
__global__ __launch_bounds__(256) void cast_scale_kernel(
    const float* __restrict__ src, __bf16* __restrict__ dst,
    const float* __restrict__ gamma_log, int rowlen, float mul, unsigned int n) {
  unsigned int i = blockIdx.x * 256u + threadIdx.x;
  if (i >= n) return;
  float v = src[i] * mul;
  if (gamma_log) v *= __expf(gamma_log[i / (unsigned)rowlen]);
  dst[i] = (__bf16)v;
}

// ---------------------------------------------------------------------------
// LayerNorm over H=512: one wave (32 lanes) per row, 8 rows per block.
// Writes bf16 (feeds the WMMA GEMMs and the D*u epilogue).
// ---------------------------------------------------------------------------
__global__ __launch_bounds__(256) void layernorm_bf16(
    const float* __restrict__ h, const float* __restrict__ w,
    const float* __restrict__ b, __bf16* __restrict__ z) {
  const int C = HH;
  int row  = blockIdx.x * 8 + (threadIdx.x >> 5);
  int lane = threadIdx.x & 31;
  const float* p = h + (size_t)row * C;
  float s = 0.f, s2 = 0.f;
  #pragma unroll
  for (int j = lane; j < C; j += 32) { float v = p[j]; s += v; s2 += v * v; }
  #pragma unroll
  for (int off = 16; off > 0; off >>= 1) {
    s  += __shfl_xor(s,  off, 32);
    s2 += __shfl_xor(s2, off, 32);
  }
  float mu   = s / C;
  float var  = s2 / C - mu * mu;      // biased variance (torch LayerNorm)
  float rstd = rsqrtf(var + 1e-5f);
  __bf16* q = z + (size_t)row * C;
  #pragma unroll
  for (int j = lane; j < C; j += 32)
    q[j] = (__bf16)((p[j] - mu) * rstd * w[j] + b[j]);
}

// ---------------------------------------------------------------------------
// WMMA GEMM:  Y[M,N] = A[M,K](bf16) * W[N,K]^T(bf16)  (+ variants)
// Block: 256 threads = 8 waves (4 along M x 2 along N).
// Wave tile: 32(M) x WN*16(N)  ->  block tile 128 x (WN*32).
//   WN=4 for the single-GEMM epilogues (FF1/FF2/proj: 8 WMMA per 6 loads),
//   WN=2 for the dual variants (which already carry 2x fragments).
// Epilogues:
//   EPI_F32       : outF = acc                       (input projection)
//   EPI_TRANSDUAL : DUAL weights (W,W2), same A -> two accs;
//                   store transposed outF/outF2[n*ldT + m] as 2x b128/lane
//   EPI_DU        : DUAL A (A,A2) x (W,W2) summed into one acc;
//                   outH = bf16(acc + Dvec[n] * U[m,n])      (LRU output)
//   EPI_GELU      : outH = bf16(gelu(acc + bias[n]))         (FF1)
//   EPI_RESID     : outF[m,n] += acc + bias[n]               (FF2 residual)
// ---------------------------------------------------------------------------
#define EPI_F32       0
#define EPI_TRANSDUAL 1
#define EPI_DU        2
#define EPI_GELU      3
#define EPI_RESID     4

template <int EPI, bool DUAL, int WN>
__global__ __launch_bounds__(256) void gemm_wmma(
    const __bf16* __restrict__ A,  const __bf16* __restrict__ A2,
    const __bf16* __restrict__ W,  const __bf16* __restrict__ W2,
    float* __restrict__ outF, float* __restrict__ outF2,
    __bf16* __restrict__ outH,
    const float* __restrict__ bias, const float* __restrict__ Dvec,
    const __bf16* __restrict__ U,
    int M, int N, int K, int ldT) {
  const int lane  = threadIdx.x & 31;
  const int wave  = threadIdx.x >> 5;
  const int wm    = wave & 3;          // 4 waves along M
  const int wn    = wave >> 2;         // 2 waves along N
  const int mbase = blockIdx.x * 128 + wm * 32;
  const int nbase = blockIdx.y * (WN * 32) + wn * (WN * 16);
  const int lhalf = lane >> 4;         // 0 | 1  (lane group)
  const int lmod  = lane & 15;

  const v8f vzero = {0.f, 0.f, 0.f, 0.f, 0.f, 0.f, 0.f, 0.f};
  v8f acc[2][WN], acc2[2][WN];
  #pragma unroll
  for (int i = 0; i < 2; ++i)
    #pragma unroll
    for (int j = 0; j < WN; ++j) { acc[i][j] = vzero; acc2[i][j] = vzero; }

  for (int k = 0; k < K; k += 32) {
    Frag a[2], a2[2], b[WN], b2[WN];
    #pragma unroll
    for (int i = 0; i < 2; ++i) {
      // A fragment (16x32): lane half 0 -> K {0..7,16..23}, half 1 -> {8..15,24..31}
      size_t arow = (size_t)(mbase + i * 16 + lmod) * K + k + lhalf * 8;
      a[i].u[0] = *(const u32x4*)(A + arow);
      a[i].u[1] = *(const u32x4*)(A + arow + 16);
      if (DUAL && EPI == EPI_DU) {
        a2[i].u[0] = *(const u32x4*)(A2 + arow);
        a2[i].u[1] = *(const u32x4*)(A2 + arow + 16);
      }
    }
    #pragma unroll
    for (int j = 0; j < WN; ++j) {
      // B fragment (32x16): column n per lane, half 0 -> K {0..7,16..23}, half 1 -> {8..15,24..31}
      size_t brow = (size_t)(nbase + j * 16 + lmod) * K + k + lhalf * 8;
      b[j].u[0] = *(const u32x4*)(W + brow);
      b[j].u[1] = *(const u32x4*)(W + brow + 16);
      if (DUAL) {
        b2[j].u[0] = *(const u32x4*)(W2 + brow);
        b2[j].u[1] = *(const u32x4*)(W2 + brow + 16);
      }
    }
    #pragma unroll
    for (int i = 0; i < 2; ++i)
      #pragma unroll
      for (int j = 0; j < WN; ++j) {
        acc[i][j] = wmma_bf16(a[i].v, b[j].v, acc[i][j]);
        if (DUAL) {
          if (EPI == EPI_TRANSDUAL)          // same A, second weight set
            acc2[i][j] = wmma_bf16(a[i].v, b2[j].v, acc2[i][j]);
          else                               // EPI_DU: A2*W2 added into acc
            acc[i][j] = wmma_bf16(a2[i].v, b2[j].v, acc[i][j]);
        }
      }
  }

  // Epilogue. C/D layout: lane<16 -> N=lmod, M=r ; lane>=16 -> N=lmod, M=8+r
  if (EPI == EPI_TRANSDUAL) {
    // Each lane owns column n and 8 consecutive rows -> 2x b128 per tile.
    #pragma unroll
    for (int i = 0; i < 2; ++i)
      #pragma unroll
      for (int j = 0; j < WN; ++j) {
        int n  = nbase + j * 16 + lmod;
        int m0 = mbase + i * 16 + 8 * lhalf;
        AccSp t;
        t.v = acc[i][j];
        *(u32x4*)(outF + (size_t)n * ldT + m0)     = t.q[0];
        *(u32x4*)(outF + (size_t)n * ldT + m0 + 4) = t.q[1];
        t.v = acc2[i][j];
        *(u32x4*)(outF2 + (size_t)n * ldT + m0)     = t.q[0];
        *(u32x4*)(outF2 + (size_t)n * ldT + m0 + 4) = t.q[1];
      }
  } else {
    #pragma unroll
    for (int i = 0; i < 2; ++i)
      #pragma unroll
      for (int j = 0; j < WN; ++j)
        #pragma unroll
        for (int r = 0; r < 8; ++r) {
          int m = mbase + i * 16 + r + 8 * lhalf;
          int n = nbase + j * 16 + lmod;
          float v = acc[i][j][r];
          if (EPI == EPI_F32) {
            outF[(size_t)m * N + n] = v;
          } else if (EPI == EPI_DU) {
            float u = (float)U[(size_t)m * N + n];
            outH[(size_t)m * N + n] = (__bf16)(v + Dvec[n] * u);
          } else if (EPI == EPI_GELU) {
            float xg = v + bias[n];                       // exact GELU
            outH[(size_t)m * N + n] =
                (__bf16)(0.5f * xg * (1.0f + erff(xg * 0.70710678118654752f)));
          } else if (EPI == EPI_RESID) {
            size_t o = (size_t)m * N + n;
            outF[o] = outF[o] + v + bias[n];
          }
        }
  }
}

// ---------------------------------------------------------------------------
// LRU scan: h_t = lam * h_{t-1} + Bu_t, complex diagonal, per (b, n).
// Data is in transposed layout [N, B*T] so each thread reads 16 contiguous
// complex elements. Block (256 threads) covers one full T=4096 sequence:
// local serial scan -> Hillis-Steele block scan of (lam^16, carry) pairs
// with combine (a1,b1)o(a2,b2) = (a1*a2, a2*b1 + b2) -> carry fix-up.
// ---------------------------------------------------------------------------
__global__ __launch_bounds__(256) void lru_scan(
    float* __restrict__ re, float* __restrict__ im,
    const float* __restrict__ nu_log, const float* __restrict__ theta_log) {
  const int n = blockIdx.x, b = blockIdx.y, tid = threadIdx.x;
  size_t base = (size_t)n * MM + (size_t)b * TT + (size_t)tid * 16;

  float mag = __expf(-__expf(nu_log[n]));
  float ang = __expf(theta_log[n]);
  float lr = mag * __cosf(ang), li = mag * __sinf(ang);

  float xr[16], xi[16];
  #pragma unroll
  for (int j = 0; j < 16; ++j) { xr[j] = re[base + j]; xi[j] = im[base + j]; }

  // local inclusive scan over 16 elements (h starts at 0)
  float hr = 0.f, hi = 0.f;
  #pragma unroll
  for (int j = 0; j < 16; ++j) {
    float nr = lr * hr - li * hi + xr[j];
    float ni = lr * hi + li * hr + xi[j];
    hr = nr; hi = ni; xr[j] = hr; xi[j] = hi;
  }
  // lam^16 via 4 complex squarings
  float ar = lr, ai = li;
  #pragma unroll
  for (int s = 0; s < 4; ++s) {
    float nr = ar * ar - ai * ai, ni = 2.f * ar * ai;
    ar = nr; ai = ni;
  }

  __shared__ float sar[256], sai[256], sbr[256], sbi[256];
  sar[tid] = ar; sai[tid] = ai; sbr[tid] = hr; sbi[tid] = hi;
  __syncthreads();
  for (int off = 1; off < 256; off <<= 1) {
    float par = 0.f, pai = 0.f, pbr = 0.f, pbi = 0.f;
    bool has = tid >= off;
    if (has) { par = sar[tid-off]; pai = sai[tid-off];
               pbr = sbr[tid-off]; pbi = sbi[tid-off]; }
    __syncthreads();
    if (has) {
      float car = sar[tid], cai = sai[tid], cbr = sbr[tid], cbi = sbi[tid];
      sar[tid] = par * car - pai * cai;
      sai[tid] = par * cai + pai * car;
      sbr[tid] = car * pbr - cai * pbi + cbr;
      sbi[tid] = car * pbi + cai * pbr + cbi;
    }
    __syncthreads();
  }
  float spr = 0.f, spi = 0.f;
  if (tid > 0) { spr = sbr[tid - 1]; spi = sbi[tid - 1]; }

  // fix-up: h_j += lam^(j+1) * carry_in ; store in place
  float pr = lr, pi = li;
  #pragma unroll
  for (int j = 0; j < 16; ++j) {
    re[base + j] = xr[j] + (pr * spr - pi * spi);
    im[base + j] = xi[j] + (pr * spi + pi * spr);
    float nr = pr * lr - pi * li, ni = pr * li + pi * lr;
    pr = nr; pi = ni;
  }
}

// ---------------------------------------------------------------------------
// Tiled transpose [rows_in, cols_in](f32) -> [cols_in, rows_in](bf16) via LDS
// ---------------------------------------------------------------------------
__global__ __launch_bounds__(256) void transpose_cast(
    const float* __restrict__ in, __bf16* __restrict__ out,
    int rows_in, int cols_in) {
  __shared__ float tile[32][33];
  int c0 = blockIdx.x * 32, r0 = blockIdx.y * 32;
  int tx = threadIdx.x & 31, ty = threadIdx.x >> 5;   // 32 x 8
  #pragma unroll
  for (int j = 0; j < 4; ++j)
    tile[ty + 8 * j][tx] = in[(size_t)(r0 + ty + 8 * j) * cols_in + c0 + tx];
  __syncthreads();
  #pragma unroll
  for (int j = 0; j < 4; ++j)
    out[(size_t)(c0 + ty + 8 * j) * rows_in + r0 + tx] =
        (__bf16)tile[tx][ty + 8 * j];
}

// ---------------------------------------------------------------------------
// Workspace layout (bytes). Peak ~153 MB; FF intermediate aliases the BuT
// region (consumed by transpose-back before FF1 runs).
// ---------------------------------------------------------------------------
static constexpr size_t SZ_HH_BF  = (size_t)HH * HH * 2;        // 0.5 MB
static constexpr size_t SZ_W1_BF  = (size_t)FF * HH * 2;        // 2 MB
static constexpr size_t SZ_LAYERW = 4 * SZ_HH_BF + 2 * SZ_W1_BF;
static constexpr size_t SZ_ACT_BF = (size_t)MM * HH * 2;        // 16 MB
static constexpr size_t SZ_ACT_F  = (size_t)MM * HH * 4;        // 32 MB

static constexpr size_t OFF_WPROJ  = 0;
static constexpr size_t OFF_LW     = OFF_WPROJ + SZ_HH_BF;
static constexpr size_t OFF_ZBF    = OFF_LW + LL * SZ_LAYERW;
static constexpr size_t OFF_BUT_RE = OFF_ZBF + SZ_ACT_BF;
static constexpr size_t OFF_BUT_IM = OFF_BUT_RE + SZ_ACT_F;
static constexpr size_t OFF_GBF    = OFF_BUT_RE;                // alias (64 MB)
static constexpr size_t OFF_HRE    = OFF_BUT_IM + SZ_ACT_F;
static constexpr size_t OFF_HIM    = OFF_HRE + SZ_ACT_BF;
static constexpr size_t OFF_YBF    = OFF_HIM + SZ_ACT_BF;

static inline __bf16* wsb(void* ws, size_t off) { return (__bf16*)((char*)ws + off); }
static inline float*  wsf(void* ws, size_t off) { return (float*)((char*)ws + off); }

extern "C" void kernel_launch(void* const* d_in, const int* in_sizes, int n_in,
                              void* d_out, int out_size, void* d_ws, size_t ws_size,
                              hipStream_t stream) {
  (void)in_sizes; (void)n_in; (void)out_size; (void)ws_size;
  const float* x         = (const float*)d_in[0];
  const float* W_proj    = (const float*)d_in[1];
  const float* ln_w      = (const float*)d_in[2];
  const float* ln_b      = (const float*)d_in[3];
  const float* nu_log    = (const float*)d_in[4];
  const float* theta_log = (const float*)d_in[5];
  const float* gamma_log = (const float*)d_in[6];
  const float* B_re      = (const float*)d_in[7];
  const float* B_im      = (const float*)d_in[8];
  const float* C_re      = (const float*)d_in[9];
  const float* C_im      = (const float*)d_in[10];
  const float* Dv        = (const float*)d_in[11];
  const float* ff_w1     = (const float*)d_in[12];
  const float* ff_b1     = (const float*)d_in[13];
  const float* ff_w2     = (const float*)d_in[14];
  const float* ff_b2     = (const float*)d_in[15];
  float* hout = (float*)d_out;   // h lives in d_out, updated in place per layer

  const dim3 blk(256);
  auto cast = [&](const float* src, size_t off, const float* gl, int rowlen,
                  float mul, size_t n) {
    cast_scale_kernel<<<(unsigned)((n + 255) / 256), blk, 0, stream>>>(
        src, wsb(d_ws, off), gl, rowlen, mul, (unsigned)n);
  };

  // --- weight casts (deterministic, redone every call) --------------------
  cast(W_proj, OFF_WPROJ, nullptr, 1, 1.f, (size_t)HH * II);
  for (int l = 0; l < LL; ++l) {
    size_t lb = OFF_LW + (size_t)l * SZ_LAYERW;
    cast(B_re + (size_t)l * HH * HH, lb + 0 * SZ_HH_BF, gamma_log + l * HH, HH, 1.f,  (size_t)HH * HH);
    cast(B_im + (size_t)l * HH * HH, lb + 1 * SZ_HH_BF, gamma_log + l * HH, HH, 1.f,  (size_t)HH * HH);
    cast(C_re + (size_t)l * HH * HH, lb + 2 * SZ_HH_BF, nullptr, 1, 1.f,  (size_t)HH * HH);
    cast(C_im + (size_t)l * HH * HH, lb + 3 * SZ_HH_BF, nullptr, 1, -1.f, (size_t)HH * HH); // negate once
    cast(ff_w1 + (size_t)l * FF * HH, lb + 4 * SZ_HH_BF,            nullptr, 1, 1.f, (size_t)FF * HH);
    cast(ff_w2 + (size_t)l * HH * FF, lb + 4 * SZ_HH_BF + SZ_W1_BF, nullptr, 1, 1.f, (size_t)HH * FF);
  }

  __bf16* zbf   = wsb(d_ws, OFF_ZBF);
  float*  butre = wsf(d_ws, OFF_BUT_RE);
  float*  butim = wsf(d_ws, OFF_BUT_IM);
  __bf16* hrebf = wsb(d_ws, OFF_HRE);
  __bf16* himbf = wsb(d_ws, OFF_HIM);
  __bf16* ybf   = wsb(d_ws, OFF_YBF);
  __bf16* gbf   = wsb(d_ws, OFF_GBF);

  const dim3 gHH2(MM / 128, HH / 64);    // dual GEMMs, WN=2: (128, 8)
  const dim3 gHH4(MM / 128, HH / 128);   // single GEMMs, WN=4: (128, 4)
  const dim3 gFF4(MM / 128, FF / 128);   // FF1, WN=4: (128, 16)

  // --- input projection: h = x @ W_proj^T ---------------------------------
  cast(x, OFF_ZBF, nullptr, 1, 1.f, (size_t)MM * II);
  gemm_wmma<EPI_F32, false, 4><<<gHH4, blk, 0, stream>>>(
      zbf, nullptr, wsb(d_ws, OFF_WPROJ), nullptr,
      hout, nullptr, nullptr, nullptr, nullptr, nullptr, MM, HH, II, 0);

  // --- layers -------------------------------------------------------------
  for (int l = 0; l < LL; ++l) {
    size_t lb = OFF_LW + (size_t)l * SZ_LAYERW;
    __bf16* brebf = wsb(d_ws, lb + 0 * SZ_HH_BF);
    __bf16* bimbf = wsb(d_ws, lb + 1 * SZ_HH_BF);
    __bf16* crebf = wsb(d_ws, lb + 2 * SZ_HH_BF);
    __bf16* cimbf = wsb(d_ws, lb + 3 * SZ_HH_BF);   // holds -C_im
    __bf16* w1bf  = wsb(d_ws, lb + 4 * SZ_HH_BF);
    __bf16* w2bf  = wsb(d_ws, lb + 4 * SZ_HH_BF + SZ_W1_BF);

    // z = LayerNorm(h)  (bf16)
    layernorm_bf16<<<MM / 8, blk, 0, stream>>>(hout, ln_w + l * HH, ln_b + l * HH, zbf);

    // Bu_re/Bu_im = z @ (B_re*e^g)^T / (B_im*e^g)^T, stored transposed [N, B*T]
    gemm_wmma<EPI_TRANSDUAL, true, 2><<<gHH2, blk, 0, stream>>>(
        zbf, nullptr, brebf, bimbf,
        butre, butim, nullptr, nullptr, nullptr, nullptr, MM, HH, HH, MM);

    // parallel complex scan along T, in place
    lru_scan<<<dim3(HH, BB), blk, 0, stream>>>(butre, butim, nu_log + l * HH, theta_log + l * HH);

    // back to row-major bf16 for the output GEMM
    transpose_cast<<<dim3(MM / 32, HH / 32), blk, 0, stream>>>(butre, hrebf, HH, MM);
    transpose_cast<<<dim3(MM / 32, HH / 32), blk, 0, stream>>>(butim, himbf, HH, MM);

    // y = h_re@C_re^T + h_im@(-C_im)^T + D*u   (bf16 out)
    gemm_wmma<EPI_DU, true, 2><<<gHH2, blk, 0, stream>>>(
        hrebf, himbf, crebf, cimbf,
        nullptr, nullptr, ybf, nullptr, Dv + l * HH, zbf, MM, HH, HH, 0);

    // g = gelu(y @ w1^T + b1)   (bf16 out)
    gemm_wmma<EPI_GELU, false, 4><<<gFF4, blk, 0, stream>>>(
        ybf, nullptr, w1bf, nullptr,
        nullptr, nullptr, gbf, ff_b1 + (size_t)l * FF, nullptr, nullptr, MM, FF, HH, 0);

    // h += g @ w2^T + b2   (fp32, in place on d_out)
    gemm_wmma<EPI_RESID, false, 4><<<gHH4, blk, 0, stream>>>(
        gbf, nullptr, w2bf, nullptr,
        hout, nullptr, nullptr, ff_b2 + (size_t)l * HH, nullptr, nullptr, MM, HH, FF, 0);
  }
}
